// Phi3MoE_6665789243464
// MI455X (gfx1250) — compile-verified, matrix operand order checked by reading, not run on hardware
//
#include <hip/hip_runtime.h>
#include <hip/hip_bf16.h>

#define TT 4096
#define HH 4096
#define II 6400
#define EE 8

typedef __attribute__((ext_vector_type(16))) __bf16 v16bf;
typedef __attribute__((ext_vector_type(8)))  float  v8f;

union FragBF { v16bf v; unsigned int u[8]; };

// Pack two f32 -> packed bf16 pair, RNE, 32-bit ops only (no true16 zext spam):
// round-bias both values, then v_perm_b32 grabs the two high halves.
__device__ __forceinline__ unsigned int pack2bf(float a, float b) {
    unsigned int ua = __float_as_uint(a), ub = __float_as_uint(b);
    ua += 0x7fffu + ((ua >> 16) & 1u);
    ub += 0x7fffu + ((ub >> 16) & 1u);
    return __builtin_amdgcn_perm(ub, ua, 0x07060302u);   // {ub[31:16], ua[31:16]}
}
__device__ __forceinline__ unsigned short f2bf(float f) {
    unsigned int u = __float_as_uint(f);
    u += 0x7fffu + ((u >> 16) & 1u);
    return (unsigned short)(u >> 16);
}

// ---------------------------------------------------------------------------
// Zero output + per-expert counters
// ---------------------------------------------------------------------------
__global__ void moe_zero(float4* __restrict__ out, int* __restrict__ counts) {
    size_t i = (size_t)blockIdx.x * blockDim.x + threadIdx.x;   // 16384*256 = T*H/4
    out[i] = make_float4(0.f, 0.f, 0.f, 0.f);
    if (blockIdx.x == 0 && threadIdx.x < EE) counts[threadIdx.x] = 0;
}

// ---------------------------------------------------------------------------
// Router: one wave (32 lanes) per token.
// ---------------------------------------------------------------------------
__global__ __launch_bounds__(256)
void moe_router(const float* __restrict__ x, const float* __restrict__ gw,
                int* __restrict__ counts, int* __restrict__ tok_idx,
                float* __restrict__ tok_scale) {
    int wave = threadIdx.x >> 5;
    int lane = threadIdx.x & 31;
    int t = blockIdx.x * 8 + wave;

    const float* xr = x + (size_t)t * HH;
    float acc[EE];
#pragma unroll
    for (int e = 0; e < EE; ++e) acc[e] = 0.f;

    for (int h = lane; h < HH; h += 32) {
        float xv = xr[h];
#pragma unroll
        for (int e = 0; e < EE; ++e) acc[e] += xv * gw[e * HH + h];
    }
#pragma unroll
    for (int e = 0; e < EE; ++e) {
#pragma unroll
        for (int off = 16; off > 0; off >>= 1)
            acc[e] += __shfl_xor(acc[e], off, 32);
    }
    if (lane == 0) {
        float mx = acc[0];
#pragma unroll
        for (int e = 1; e < EE; ++e) mx = fmaxf(mx, acc[e]);
        float p[EE], s = 0.f;
#pragma unroll
        for (int e = 0; e < EE; ++e) { p[e] = __expf(acc[e] - mx); s += p[e]; }
        int i0 = 0;
#pragma unroll
        for (int e = 1; e < EE; ++e) if (p[e] > p[i0]) i0 = e;
        int i1 = -1;
#pragma unroll
        for (int e = 0; e < EE; ++e) {
            if (e == i0) continue;
            if (i1 < 0 || p[e] > p[i1]) i1 = e;
        }
        float v0 = p[i0] / s, v1 = p[i1] / s;
        float inv = 1.f / (v0 + v1);
        v0 *= inv; v1 *= inv;
        int p0 = atomicAdd(&counts[i0], 1);
        tok_idx[i0 * TT + p0] = t;  tok_scale[i0 * TT + p0] = v0;
        int p1 = atomicAdd(&counts[i1], 1);
        tok_idx[i1 * TT + p1] = t;  tok_scale[i1 * TT + p1] = v1;
    }
}

// ---------------------------------------------------------------------------
// Fragment gather from LDS (bf16, stride 34 halfwords). ISA 16-bit A layout.
// ---------------------------------------------------------------------------
__device__ __forceinline__ void load_frag(FragBF& f, const unsigned short* rowp, int fh) {
#pragma unroll
    for (int j = 0; j < 8; ++j) {
        int k = ((j & 4) << 2) + (fh << 3) + ((j & 3) << 1);
        f.u[j] = *(const unsigned int*)(rowp + k);
    }
}

// ---------------------------------------------------------------------------
// GEMM1 (per expert): h = silu(Xg@w1^T) * (Xg@w3^T) -> bf16.
// Double-buffered LDS, register-staged global loads. Block 128x64, 8 waves 32x32.
// Grid: x = M-tiles (fast -> B-chunk L2 reuse), y = N-tiles.
// ---------------------------------------------------------------------------
__global__ __launch_bounds__(256)
void moe_gemm1(const float* __restrict__ x, const float* __restrict__ w1,
               const float* __restrict__ w3, const int* __restrict__ counts,
               const int* __restrict__ tok_idx, unsigned short* __restrict__ hbuf,
               int e) {
    const int count = counts[e];
    const int m0 = blockIdx.x * 128;
    if (m0 >= count) return;
    const int n0 = blockIdx.y * 64;

    const float* W1 = w1 + (size_t)e * II * HH;
    const float* W3 = w3 + (size_t)e * II * HH;
    const int* list = tok_idx + e * TT;

    __shared__ unsigned short Asm[2][128][34];
    __shared__ unsigned short B1s[2][64][34];
    __shared__ unsigned short B3s[2][64][34];

    const int tid = threadIdx.x;
    const int arow = tid >> 1;
    const int acol = (tid & 1) * 16;
    int lrow = m0 + arow; if (lrow > count - 1) lrow = count - 1;
    const float* aptr = x + (size_t)list[lrow] * HH + acol;
    const int brow = tid >> 2;
    const int bcol = (tid & 3) * 8;
    const float* b1ptr = W1 + (size_t)(n0 + brow) * HH + bcol;
    const float* b3ptr = W3 + (size_t)(n0 + brow) * HH + bcol;

    const int wid = tid >> 5, lane = tid & 31;
    const int wm = (wid & 3) * 32;
    const int wn = (wid >> 2) * 32;
    const int fm = lane & 15, fh = lane >> 4;

    v8f accG[2][2] = {}, accU[2][2] = {};
    float4 aR[4], b1R[2], b3R[2];

    // ---- prologue: stage 0 -> buf 0
#pragma unroll
    for (int i = 0; i < 4; ++i) aR[i] = ((const float4*)aptr)[i];
#pragma unroll
    for (int i = 0; i < 2; ++i) {
        b1R[i] = ((const float4*)b1ptr)[i];
        b3R[i] = ((const float4*)b3ptr)[i];
    }
#pragma unroll
    for (int i = 0; i < 4; ++i) {
        *(unsigned int*)&Asm[0][arow][acol + i * 4 + 0] = pack2bf(aR[i].x, aR[i].y);
        *(unsigned int*)&Asm[0][arow][acol + i * 4 + 2] = pack2bf(aR[i].z, aR[i].w);
    }
#pragma unroll
    for (int i = 0; i < 2; ++i) {
        *(unsigned int*)&B1s[0][brow][bcol + i * 4 + 0] = pack2bf(b1R[i].x, b1R[i].y);
        *(unsigned int*)&B1s[0][brow][bcol + i * 4 + 2] = pack2bf(b1R[i].z, b1R[i].w);
        *(unsigned int*)&B3s[0][brow][bcol + i * 4 + 0] = pack2bf(b3R[i].x, b3R[i].y);
        *(unsigned int*)&B3s[0][brow][bcol + i * 4 + 2] = pack2bf(b3R[i].z, b3R[i].w);
    }
    __syncthreads();

    int p = 0;
    for (int kk = 0; kk < HH; kk += 32) {
        const bool more = (kk + 32) < HH;
        if (more) {                               // stage k+1 -> registers (latency hidden by WMMAs)
#pragma unroll
            for (int i = 0; i < 4; ++i) aR[i] = ((const float4*)(aptr + kk + 32))[i];
#pragma unroll
            for (int i = 0; i < 2; ++i) {
                b1R[i] = ((const float4*)(b1ptr + kk + 32))[i];
                b3R[i] = ((const float4*)(b3ptr + kk + 32))[i];
            }
            if (kk + 64 < HH) {
                __builtin_prefetch(b1ptr + kk + 64, 0, 1);
                __builtin_prefetch(b3ptr + kk + 64, 0, 1);
            }
        }
        // ---- compute stage k from buf p
        FragBF a[2], b1f[2], b3f[2];
#pragma unroll
        for (int mi = 0; mi < 2; ++mi) load_frag(a[mi], &Asm[p][wm + mi * 16 + fm][0], fh);
#pragma unroll
        for (int ni = 0; ni < 2; ++ni) {
            load_frag(b1f[ni], &B1s[p][wn + ni * 16 + fm][0], fh);
            load_frag(b3f[ni], &B3s[p][wn + ni * 16 + fm][0], fh);
        }
#pragma unroll
        for (int mi = 0; mi < 2; ++mi)
#pragma unroll
            for (int ni = 0; ni < 2; ++ni) {
                accG[mi][ni] = __builtin_amdgcn_wmma_f32_16x16x32_bf16(
                    false, a[mi].v, false, b1f[ni].v, (short)0, accG[mi][ni], false, false);
                accU[mi][ni] = __builtin_amdgcn_wmma_f32_16x16x32_bf16(
                    false, a[mi].v, false, b3f[ni].v, (short)0, accU[mi][ni], false, false);
            }
        if (more) {                               // stage k+1 -> buf 1-p
            const int q = p ^ 1;
#pragma unroll
            for (int i = 0; i < 4; ++i) {
                *(unsigned int*)&Asm[q][arow][acol + i * 4 + 0] = pack2bf(aR[i].x, aR[i].y);
                *(unsigned int*)&Asm[q][arow][acol + i * 4 + 2] = pack2bf(aR[i].z, aR[i].w);
            }
#pragma unroll
            for (int i = 0; i < 2; ++i) {
                *(unsigned int*)&B1s[q][brow][bcol + i * 4 + 0] = pack2bf(b1R[i].x, b1R[i].y);
                *(unsigned int*)&B1s[q][brow][bcol + i * 4 + 2] = pack2bf(b1R[i].z, b1R[i].w);
                *(unsigned int*)&B3s[q][brow][bcol + i * 4 + 0] = pack2bf(b3R[i].x, b3R[i].y);
                *(unsigned int*)&B3s[q][brow][bcol + i * 4 + 2] = pack2bf(b3R[i].z, b3R[i].w);
            }
            __syncthreads();
            p = q;
        }
    }

    const int rbase = fh * 8;
#pragma unroll
    for (int mi = 0; mi < 2; ++mi)
#pragma unroll
        for (int ni = 0; ni < 2; ++ni)
#pragma unroll
            for (int r = 0; r < 8; ++r) {
                int row = m0 + wm + mi * 16 + rbase + r;
                int col = n0 + wn + ni * 16 + fm;
                float g = accG[mi][ni][r];
                float u = accU[mi][ni][r];
                float hv = (g / (1.f + __expf(-g))) * u;   // silu(g) * u
                hbuf[(size_t)row * II + col] = f2bf(hv);
            }
}

// ---------------------------------------------------------------------------
// GEMM2 (per expert): out[tok] += scale * (h @ w2^T). Block 128x128.
// Double-buffered; A tile via CDNA5 async global->LDS DMA (ASYNCcnt),
// B tile register-staged fp32->bf16. Grid: x = M-tiles, y = N-tiles.
// ---------------------------------------------------------------------------
__global__ __launch_bounds__(256)
void moe_gemm2(const unsigned short* __restrict__ hbuf, const float* __restrict__ w2,
               const int* __restrict__ counts, const int* __restrict__ tok_idx,
               const float* __restrict__ tok_scale, float* __restrict__ out, int e) {
    const int count = counts[e];
    const int m0 = blockIdx.x * 128;
    if (m0 >= count) return;
    const int n0 = blockIdx.y * 128;

    const float* W2 = w2 + (size_t)e * HH * II;
    const int*   lidx = tok_idx  + e * TT;
    const float* lscl = tok_scale + e * TT;

    __shared__ unsigned short Asm[2][128][34];
    __shared__ unsigned short Bsm[2][128][34];

    const int tid = threadIdx.x;
    const int arow = tid >> 1;
    const int acol = (tid & 1) * 16;                 // halfwords
    const unsigned short* aptr = hbuf + (size_t)(m0 + arow) * II + acol;
    const float* bptr = W2 + (size_t)(n0 + arow) * II + acol;

#if defined(__gfx1250__)
    const unsigned int ldsA[2] = {
        (unsigned int)(unsigned long long)&Asm[0][arow][acol],
        (unsigned int)(unsigned long long)&Asm[1][arow][acol]
    };
#endif

    const int wid = tid >> 5, lane = tid & 31;
    const int wm = (wid & 3) * 32;
    const int wn = (wid >> 2) * 64;
    const int fm = lane & 15, fh = lane >> 4;

    v8f acc[2][4] = {};
    float4 bR[4];

    // ---- prologue: stage 0 -> buf 0
#if defined(__gfx1250__)
    {
        const unsigned short* g0 = aptr;
        asm volatile(
            "global_load_async_to_lds_b128 %0, %2, off\n\t"
            "global_load_async_to_lds_b128 %1, %3, off"
            :: "v"(ldsA[0]), "v"(ldsA[0] + 16u), "v"(g0), "v"(g0 + 8)
            : "memory");
    }
#else
    {
        const uint4* ap = (const uint4*)aptr;
        uint4 a0 = ap[0], a1 = ap[1];
        unsigned int* as = (unsigned int*)&Asm[0][arow][acol];
        as[0] = a0.x; as[1] = a0.y; as[2] = a0.z; as[3] = a0.w;
        as[4] = a1.x; as[5] = a1.y; as[6] = a1.z; as[7] = a1.w;
    }
#endif
#pragma unroll
    for (int i = 0; i < 4; ++i) bR[i] = ((const float4*)bptr)[i];
#pragma unroll
    for (int i = 0; i < 4; ++i) {
        *(unsigned int*)&Bsm[0][arow][acol + i * 4 + 0] = pack2bf(bR[i].x, bR[i].y);
        *(unsigned int*)&Bsm[0][arow][acol + i * 4 + 2] = pack2bf(bR[i].z, bR[i].w);
    }
#if defined(__gfx1250__)
    asm volatile("s_wait_asynccnt 0x0" ::: "memory");
#endif
    __syncthreads();

    int p = 0;
    for (int kk = 0; kk < II; kk += 32) {
        const bool more = (kk + 32) < II;
        if (more) {
            // next A tile: async DMA straight into the other LDS buffer
#if defined(__gfx1250__)
            {
                const unsigned short* g0 = aptr + kk + 32;
                unsigned int l0 = ldsA[p ^ 1];
                asm volatile(
                    "global_load_async_to_lds_b128 %0, %2, off\n\t"
                    "global_load_async_to_lds_b128 %1, %3, off"
                    :: "v"(l0), "v"(l0 + 16u), "v"(g0), "v"(g0 + 8)
                    : "memory");
            }
#else
            {
                const uint4* ap = (const uint4*)(aptr + kk + 32);
                uint4 a0 = ap[0], a1 = ap[1];
                unsigned int* as = (unsigned int*)&Asm[p ^ 1][arow][acol];
                as[0] = a0.x; as[1] = a0.y; as[2] = a0.z; as[3] = a0.w;
                as[4] = a1.x; as[5] = a1.y; as[6] = a1.z; as[7] = a1.w;
            }
#endif
#pragma unroll
            for (int i = 0; i < 4; ++i) bR[i] = ((const float4*)(bptr + kk + 32))[i];
            if (kk + 64 < II) __builtin_prefetch(bptr + kk + 64, 0, 1);
        }
        // ---- compute stage k from buf p
        FragBF a[2], b[4];
#pragma unroll
        for (int mi = 0; mi < 2; ++mi) load_frag(a[mi], &Asm[p][wm + mi * 16 + fm][0], fh);
#pragma unroll
        for (int ni = 0; ni < 4; ++ni) load_frag(b[ni], &Bsm[p][wn + ni * 16 + fm][0], fh);
#pragma unroll
        for (int mi = 0; mi < 2; ++mi)
#pragma unroll
            for (int ni = 0; ni < 4; ++ni)
                acc[mi][ni] = __builtin_amdgcn_wmma_f32_16x16x32_bf16(
                    false, a[mi].v, false, b[ni].v, (short)0, acc[mi][ni], false, false);
        if (more) {
            const int q = p ^ 1;
#pragma unroll
            for (int i = 0; i < 4; ++i) {
                *(unsigned int*)&Bsm[q][arow][acol + i * 4 + 0] = pack2bf(bR[i].x, bR[i].y);
                *(unsigned int*)&Bsm[q][arow][acol + i * 4 + 2] = pack2bf(bR[i].z, bR[i].w);
            }
#if defined(__gfx1250__)
            asm volatile("s_wait_asynccnt 0x0" ::: "memory");
#endif
            __syncthreads();
            p = q;
        }
    }

    const int rbase = fh * 8;
#pragma unroll
    for (int mi = 0; mi < 2; ++mi)
#pragma unroll
        for (int r = 0; r < 8; ++r) {
            int row = m0 + wm + mi * 16 + rbase + r;
            if (row < count) {
                int tok = lidx[row];
                float s = lscl[row];
                float* orow = out + (size_t)tok * HH;
#pragma unroll
                for (int ni = 0; ni < 4; ++ni) {
                    int col = n0 + wn + ni * 16 + fm;
                    orow[col] += s * acc[mi][ni][r];
                }
            }
        }
}

// ---------------------------------------------------------------------------
extern "C" void kernel_launch(void* const* d_in, const int* in_sizes, int n_in,
                              void* d_out, int out_size, void* d_ws, size_t ws_size,
                              hipStream_t stream) {
    (void)in_sizes; (void)n_in; (void)out_size; (void)ws_size;
    const float* x      = (const float*)d_in[0];
    const float* gate_w = (const float*)d_in[1];
    const float* w1     = (const float*)d_in[2];
    const float* w3     = (const float*)d_in[3];
    const float* w2     = (const float*)d_in[4];
    float* out = (float*)d_out;

    char* ws = (char*)d_ws;
    int*   counts    = (int*)ws;                                  // 8 ints
    int*   tok_idx   = (int*)(ws + 256);                          // E*T ints
    float* tok_scale = (float*)(ws + 256 + (size_t)EE * TT * 4);  // E*T floats
    unsigned short* hbuf = (unsigned short*)(ws + (1 << 19));     // T*I bf16 (52 MB)

    moe_zero<<<(TT * (size_t)HH) / 1024, 256, 0, stream>>>((float4*)out, counts);
    moe_router<<<TT / 8, 256, 0, stream>>>(x, gate_w, counts, tok_idx, tok_scale);

    for (int e = 0; e < EE; ++e) {
        dim3 g1(TT / 128, II / 64);     // x = M-tiles (fast), y = N-tiles
        moe_gemm1<<<g1, 256, 0, stream>>>(x, w1, w3, counts, tok_idx, hbuf, e);
        dim3 g2(TT / 128, HH / 128);
        moe_gemm2<<<g2, 256, 0, stream>>>(hbuf, w2, counts, tok_idx, tok_scale, out, e);
    }
}
